// MixedSparseSingleLayerWithGate_18854906429963
// MI455X (gfx1250) — compile-verified, matrix operand order checked by reading, not run on hardware
//
#include <hip/hip_runtime.h>

// ---------------------------------------------------------------------------
// MI455X (gfx1250) single transformer layer with LoRA, bf16 WMMA pipeline.
// Wave32, V_WMMA_F32_16X16X32_BF16 for all GEMMs and attention.
// GEMM tile staging uses the Tensor Data Mover (tensor_load_to_lds +
// s_wait_tensorcnt) so the VALU/VMEM pipes stay free for WMMA fragment feed.
// ---------------------------------------------------------------------------

typedef unsigned short u16;
typedef __attribute__((ext_vector_type(16))) __bf16 v16bf;
typedef __attribute__((ext_vector_type(8)))  float  v8f;
typedef __attribute__((ext_vector_type(4)))  unsigned v4u;
typedef __attribute__((ext_vector_type(8)))  int     v8i;
typedef __attribute__((ext_vector_type(4)))  int     v4i;

union FragU { uint4 u[2]; v16bf v; };

#define Bsz 4
#define Ssz 1024
#define Dsz 2048
#define Hsz 16
#define HDsz 128
#define FFsz 8192
#define Rsz 16
#define Msz (Bsz * Ssz)   // 4096 rows of activations

__device__ __forceinline__ u16 f2bf(float f) {
    unsigned u = __builtin_bit_cast(unsigned, f);
    u += 0x7fffu + ((u >> 16) & 1u);           // round-to-nearest-even
    return (u16)(u >> 16);
}
__device__ __forceinline__ float bf2f(u16 h) {
    unsigned u = ((unsigned)h) << 16;
    return __builtin_bit_cast(float, u);
}

// ---------------------------------------------------------------------------
// TDM descriptor (D#) builder + issue: load a tile_h x tile_w (elements,
// 2 bytes each) tile whose origin is gptr, row stride tensor_w elements,
// into LDS at byte offset lds_off. LDS padding: 4 DWORDs (16B) appended per
// 32 DWORDs (one 64-elem bf16 row)  =>  72-element padded LDS row stride.
// CDNA5 ISA; clang-23 6-arg builtin
// (g0, g1, g2, g3, extra int32x8 group, cpol).
// ---------------------------------------------------------------------------
__device__ __forceinline__ void tdm_load_tile_bf16(
    unsigned lds_off, const u16* gptr, unsigned tensor_w, unsigned tensor_h,
    unsigned tile_w, unsigned tile_h)
{
    unsigned long long ga = (unsigned long long)(const void*)gptr;
    v4u g0;
    g0[0] = 1u;                                   // count=1, user descriptor
    g0[1] = lds_off;                              // lds_addr  [63:32]
    g0[2] = (unsigned)ga;                         // global_addr lo
    g0[3] = (unsigned)((ga >> 32) & 0x01ffffffu)  // global_addr[56:32]
          | (2u << 30);                           // type=2 ("image")
    v8i g1;
    g1[0] = (1 << 16)                             // data_size = 2 bytes
          | (1 << 20)                             // pad_enable
          | (4 << 22)                             // pad_interval: 32 DWORDs
          | (3 << 25);                            // pad_amount:   4 DWORDs
    g1[1] = (int)((tensor_w & 0xffffu) << 16);    // tensor_dim0 lo16
    g1[2] = (int)(((tensor_w >> 16) & 0xffffu) | ((tensor_h & 0xffffu) << 16));
    g1[3] = (int)(((tensor_h >> 16) & 0xffffu) | ((tile_w & 0xffffu) << 16));
    g1[4] = (int)(tile_h & 0xffffu);              // tile_dim1 (tile_dim2 = 0)
    g1[5] = (int)tensor_w;                        // tensor_dim0_stride lo32
    g1[6] = 0;
    g1[7] = 0;
    v4i gz4 = {0, 0, 0, 0};                       // groups 2/3 unused (2D)
    v8i gz8 = {0, 0, 0, 0, 0, 0, 0, 0};           // extra group (unused)
    __builtin_amdgcn_tensor_load_to_lds(g0, g1, gz4, gz4, gz8, 0);
}

// ---------------------------------------------------------------------------
// Weight prep: Wt[n][k] = bf16( W[k][n] + sum_r A[k][r]*B[r][n] )  (LoRA fold
// + transpose so WMMA B-fragments read contiguous rows).
// ---------------------------------------------------------------------------
__global__ __launch_bounds__(256) void prep_weight_kernel(
    const float* __restrict__ W, const float* __restrict__ La,
    const float* __restrict__ Lb, u16* __restrict__ Wt, int K, int N)
{
    __shared__ float sA[32][16];   // k-tile x r
    __shared__ float sB[16][32];   // r x n-tile
    __shared__ u16   sT[32][33];   // k x n result (padded)
    const int tid = threadIdx.x;
    const int k0 = blockIdx.y * 32, n0 = blockIdx.x * 32;
    #pragma unroll
    for (int i = 0; i < 2; ++i) {
        int idx = tid + i * 256;           // 0..511
        sA[idx >> 4][idx & 15]  = La[(size_t)(k0 + (idx >> 4)) * Rsz + (idx & 15)];
        sB[idx >> 5][idx & 31]  = Lb[(size_t)(idx >> 5) * N + n0 + (idx & 31)];
    }
    __syncthreads();
    const int n = tid & 31, kb = tid >> 5;
    #pragma unroll
    for (int j = 0; j < 4; ++j) {
        int k = kb + j * 8;
        float acc = W[(size_t)(k0 + k) * N + n0 + n];
        #pragma unroll
        for (int r = 0; r < Rsz; ++r) acc += sA[k][r] * sB[r][n];
        sT[k][n] = f2bf(acc);
    }
    __syncthreads();
    #pragma unroll
    for (int j = 0; j < 4; ++j) {
        int nr = kb + j * 8, kk = tid & 31;
        Wt[(size_t)(n0 + nr) * K + k0 + kk] = sT[kk][nr];   // coalesced over kk
    }
}

// ---------------------------------------------------------------------------
// RMSNorm over D=2048, fp32 in -> bf16 out.
// ---------------------------------------------------------------------------
__global__ __launch_bounds__(256) void rmsnorm_kernel(
    const float* __restrict__ x, const float* __restrict__ w,
    u16* __restrict__ out)
{
    __shared__ float red[256];
    const int tid = threadIdx.x;
    const size_t row = (size_t)blockIdx.x * Dsz;
    float v[8], s = 0.f;
    #pragma unroll
    for (int i = 0; i < 8; ++i) { v[i] = x[row + tid + i * 256]; s += v[i] * v[i]; }
    red[tid] = s;
    __syncthreads();
    for (int off = 128; off > 0; off >>= 1) {
        if (tid < off) red[tid] += red[tid + off];
        __syncthreads();
    }
    const float inv = rsqrtf(red[0] * (1.0f / Dsz) + 1e-5f);
    #pragma unroll
    for (int i = 0; i < 8; ++i)
        out[row + tid + i * 256] = f2bf(v[i] * inv * w[tid + i * 256]);
}

// ---------------------------------------------------------------------------
// RoPE: bf16 in/out, rotate-half with cos/sin tables and position ids.
// ---------------------------------------------------------------------------
__global__ __launch_bounds__(256) void rope_kernel(
    const u16* __restrict__ in, u16* __restrict__ out,
    const float* __restrict__ cosT, const float* __restrict__ sinT,
    const int* __restrict__ pos)
{
    const size_t i = (size_t)blockIdx.x * 256 + threadIdx.x;   // over B*S*D
    const int hd = (int)(i & (HDsz - 1));
    const size_t bs = i >> 11;                                 // (b*S+s), D=2048
    const int p = pos[bs];
    const float c  = cosT[(size_t)p * HDsz + hd];
    const float sn = sinT[(size_t)p * HDsz + hd];
    const float x0 = bf2f(in[i]);
    const float xr = (hd < 64) ? -bf2f(in[i + 64]) : bf2f(in[i - 64]);
    out[i] = f2bf(x0 * c + xr * sn);
}

// ---------------------------------------------------------------------------
// Main GEMM: C[M,N] = A[M,K](bf16) * Bt[N,K](bf16)^T, fp32 accumulate.
// Block tile 128x128, 8 waves (256 thr), wave tile 32x64, K-step 64.
// Tile staging: TDM (tensor_load_to_lds), double buffered, issued by wave 0,
// overlapped with WMMA compute; synchronized with s_wait_tensorcnt+barriers.
// EPI: 0 fp32, 1 fp32+resid, 2 bf16, 3 bf16(up*silu(g)).
// ---------------------------------------------------------------------------
#define LTS 72   // padded LDS row stride (64 + 8) in bf16 elements (TDM pad)

template<int EPI>
__global__ __launch_bounds__(256) void gemm_bf16_kernel(
    const u16* __restrict__ A, const u16* __restrict__ Bt,
    float* __restrict__ Cf, u16* __restrict__ Cb,
    const float* __restrict__ resid, const u16* __restrict__ gaux,
    int M, int N, int K)
{
    __shared__ __align__(16) u16 smA[2 * 128 * LTS];
    __shared__ __align__(16) u16 smB[2 * 128 * LTS];
    const int tid = threadIdx.x;
    const int lane = tid & 31, wave = tid >> 5;
    const int half = lane >> 4, l16 = lane & 15;
    const int wm = (wave & 3) * 32, wn = (wave >> 2) * 64;
    const int m0 = blockIdx.y * 128, n0 = blockIdx.x * 128;

    v8f acc[2][4];
    #pragma unroll
    for (int i = 0; i < 2; ++i)
        #pragma unroll
        for (int j = 0; j < 4; ++j) acc[i][j] = {};

    const unsigned ldsA = (unsigned)(size_t)&smA[0];
    const unsigned ldsB = (unsigned)(size_t)&smB[0];
    const unsigned bufBytes = 128 * LTS * 2;
    const u16* gA = A  + (size_t)m0 * K;   // tile origins (k advances per step)
    const u16* gB = Bt + (size_t)n0 * K;

    if (wave == 0) {                       // prologue: DMA first tiles
        tdm_load_tile_bf16(ldsA, gA, K, M, 64, 128);
        tdm_load_tile_bf16(ldsB, gB, K, N, 64, 128);
        __builtin_amdgcn_s_wait_tensorcnt((unsigned short)0);
    }
    __syncthreads();

    const int nt = K >> 6;
    for (int t = 0; t < nt; ++t) {
        if (t + 1 < nt && wave == 0) {     // async DMA of next tiles
            tdm_load_tile_bf16(ldsA + ((t + 1) & 1) * bufBytes,
                               gA + (size_t)(t + 1) * 64, K, M, 64, 128);
            tdm_load_tile_bf16(ldsB + ((t + 1) & 1) * bufBytes,
                               gB + (size_t)(t + 1) * 64, K, N, 64, 128);
        }
        const u16* sa = &smA[(t & 1) * 128 * LTS];
        const u16* sb = &smB[(t & 1) * 128 * LTS];
        #pragma unroll
        for (int ks = 0; ks < 2; ++ks) {
            FragU a[2], b[4];
            #pragma unroll
            for (int i = 0; i < 2; ++i) {
                const u16* p = sa + (wm + i * 16 + l16) * LTS + ks * 32 + half * 8;
                a[i].u[0] = *(const uint4*)p;
                a[i].u[1] = *(const uint4*)(p + 16);
            }
            #pragma unroll
            for (int j = 0; j < 4; ++j) {
                const u16* p = sb + (wn + j * 16 + l16) * LTS + ks * 32 + half * 8;
                b[j].u[0] = *(const uint4*)p;
                b[j].u[1] = *(const uint4*)(p + 16);
            }
            #pragma unroll
            for (int i = 0; i < 2; ++i)
                #pragma unroll
                for (int j = 0; j < 4; ++j)
                    acc[i][j] = __builtin_amdgcn_wmma_f32_16x16x32_bf16(
                        false, a[i].v, false, b[j].v, (short)0, acc[i][j],
                        false, false);
        }
        __syncthreads();                   // all waves done reading buf t
        if (t + 1 < nt) {
            if (wave == 0)
                __builtin_amdgcn_s_wait_tensorcnt((unsigned short)0);
            __syncthreads();               // next buffer published
        }
    }

    #pragma unroll
    for (int i = 0; i < 2; ++i) {
        #pragma unroll
        for (int j = 0; j < 4; ++j) {
            #pragma unroll
            for (int v = 0; v < 8; ++v) {
                const int gm = m0 + wm + i * 16 + v + half * 8;
                const int gn = n0 + wn + j * 16 + l16;
                const size_t idx = (size_t)gm * N + gn;
                const float val = acc[i][j][v];
                if (EPI == 0)      Cf[idx] = val;
                else if (EPI == 1) Cf[idx] = val + resid[idx];
                else if (EPI == 2) Cb[idx] = f2bf(val);
                else {
                    const float g = bf2f(gaux[idx]);
                    Cb[idx] = f2bf(val * g / (1.f + __expf(-g)));   // up*silu(gate)
                }
            }
        }
    }
}

// ---------------------------------------------------------------------------
// Flash attention, causal, bf16 WMMA. Grid (S/64, H, B); 4 waves/block,
// each wave owns 16 query rows; V staged transposed in LDS; P re-laid-out
// through LDS into a WMMA A-fragment. Online softmax via 16-lane shfl_xor.
// ---------------------------------------------------------------------------
__global__ __launch_bounds__(128) void flash_attn_kernel(
    const u16* __restrict__ Q, const u16* __restrict__ Kc,
    const u16* __restrict__ V, u16* __restrict__ O)
{
    __shared__ __align__(16) u16 sVt[128 * 40];     // [hd][key], padded
    __shared__ __align__(16) u16 sP[4 * 16 * 40];   // per-wave P tiles
    const int tid = threadIdx.x;
    const int lane = tid & 31, wave = tid >> 5;
    const int half = lane >> 4, l16 = lane & 15;
    const int b = blockIdx.z, h = blockIdx.y, qt = blockIdx.x;
    const int q0 = qt * 64 + wave * 16;
    const float scale = 0.08838834764831845f;       // HD^-0.5

    FragU qf[4];
    {
        const u16* qbase = Q + ((size_t)(b * Ssz + q0 + l16) * Hsz + h) * HDsz;
        #pragma unroll
        for (int c = 0; c < 4; ++c) {
            const u16* p = qbase + c * 32 + half * 8;
            qf[c].u[0] = *(const uint4*)p;
            qf[c].u[1] = *(const uint4*)(p + 16);
        }
    }

    v8f oacc[8];
    #pragma unroll
    for (int t = 0; t < 8; ++t) oacc[t] = {};
    float mrun[8], lrun[8];
    #pragma unroll
    for (int v = 0; v < 8; ++v) { mrun[v] = -1e30f; lrun[v] = 0.f; }

    const int nChunk = (qt + 1) * 2;                // keys 0..qt*64+63, 32/chunk
    for (int ch = 0; ch < nChunk; ++ch) {
        const int kk0 = ch * 32;
        __syncthreads();                            // sVt reuse guard
        #pragma unroll
        for (int i = 0; i < 4; ++i) {               // stage 32 keys x 128 hd, transposed
            const int idx = tid + i * 128;          // uint4 index 0..511
            const int key = idx >> 4, seg = idx & 15;
            uint4 d = *(const uint4*)(V + ((size_t)(b * Ssz + kk0 + key) * Hsz + h) * HDsz + seg * 8);
            const u16* e = (const u16*)&d;
            #pragma unroll
            for (int j = 0; j < 8; ++j) sVt[(seg * 8 + j) * 40 + key] = e[j];
        }
        __syncthreads();

        if (kk0 <= q0 + 15) {                       // skip fully-masked chunks
            v8f sacc[2];
            #pragma unroll
            for (int s2 = 0; s2 < 2; ++s2) {
                v8f a = {};
                const u16* kbase = Kc + ((size_t)(b * Ssz + kk0 + s2 * 16 + l16) * Hsz + h) * HDsz;
                #pragma unroll
                for (int c = 0; c < 4; ++c) {
                    FragU kf;
                    const u16* p = kbase + c * 32 + half * 8;
                    kf.u[0] = *(const uint4*)p;
                    kf.u[1] = *(const uint4*)(p + 16);
                    a = __builtin_amdgcn_wmma_f32_16x16x32_bf16(
                        false, qf[c].v, false, kf.v, (short)0, a, false, false);
                }
                sacc[s2] = a;
            }
            float p0[8], p1[8];
            #pragma unroll
            for (int v = 0; v < 8; ++v) {
                const int qrow = q0 + v + half * 8;
                float s0 = sacc[0][v] * scale + (((kk0 + l16) > qrow) ? -1e9f : 0.f);
                float s1 = sacc[1][v] * scale + (((kk0 + 16 + l16) > qrow) ? -1e9f : 0.f);
                float m = fmaxf(s0, s1);
                #pragma unroll
                for (int off = 1; off < 16; off <<= 1) m = fmaxf(m, __shfl_xor(m, off));
                const float nm = fmaxf(mrun[v], m);
                const float alpha = __expf(mrun[v] - nm);
                mrun[v] = nm;
                const float e0 = __expf(s0 - nm), e1 = __expf(s1 - nm);
                p0[v] = e0; p1[v] = e1;
                float rs = e0 + e1;
                #pragma unroll
                for (int off = 1; off < 16; off <<= 1) rs += __shfl_xor(rs, off);
                lrun[v] = lrun[v] * alpha + rs;
                #pragma unroll
                for (int t = 0; t < 8; ++t) oacc[t][v] *= alpha;
            }
            u16* pw = &sP[wave * 16 * 40];
            #pragma unroll
            for (int v = 0; v < 8; ++v) {           // C-layout -> row-major P
                pw[(v + half * 8) * 40 + l16]      = f2bf(p0[v]);
                pw[(v + half * 8) * 40 + 16 + l16] = f2bf(p1[v]);
            }
            FragU pf;
            {
                const u16* p = pw + l16 * 40 + half * 8;
                pf.u[0] = *(const uint4*)p;
                pf.u[1] = *(const uint4*)(p + 16);
            }
            #pragma unroll
            for (int t = 0; t < 8; ++t) {
                FragU vf;
                const u16* p = &sVt[(t * 16 + l16) * 40 + half * 8];
                vf.u[0] = *(const uint4*)p;
                vf.u[1] = *(const uint4*)(p + 16);
                oacc[t] = __builtin_amdgcn_wmma_f32_16x16x32_bf16(
                    false, pf.v, false, vf.v, (short)0, oacc[t], false, false);
            }
        }
    }

    #pragma unroll
    for (int v = 0; v < 8; ++v) {
        const float inv = 1.f / lrun[v];
        const int qrow = q0 + v + half * 8;
        #pragma unroll
        for (int t = 0; t < 8; ++t)
            O[((size_t)(b * Ssz + qrow) * Hsz + h) * HDsz + t * 16 + l16] =
                f2bf(oacc[t][v] * inv);
    }
}

// ---------------------------------------------------------------------------
// Host-side orchestration. Workspace layout (~400 MiB), rebuilt every call.
// ---------------------------------------------------------------------------
extern "C" void kernel_launch(void* const* d_in, const int* in_sizes, int n_in,
                              void* d_out, int out_size, void* d_ws, size_t ws_size,
                              hipStream_t stream)
{
    (void)in_sizes; (void)n_in; (void)out_size; (void)ws_size;
    const float* x    = (const float*)d_in[0];
    const float* nw1  = (const float*)d_in[1];
    const float* nw2  = (const float*)d_in[2];
    const int*   pos  = (const int*)  d_in[3];
    const float* cosT = (const float*)d_in[4];
    const float* sinT = (const float*)d_in[5];
    // d_in[6] attention_mask: causal triu(-1e9) — implemented analytically.
    const float* wq = (const float*)d_in[7],  *wqa = (const float*)d_in[8],  *wqb = (const float*)d_in[9];
    const float* wk = (const float*)d_in[10], *wka = (const float*)d_in[11], *wkb = (const float*)d_in[12];
    const float* wv = (const float*)d_in[13], *wva = (const float*)d_in[14], *wvb = (const float*)d_in[15];
    const float* wo = (const float*)d_in[16], *woa = (const float*)d_in[17], *wob = (const float*)d_in[18];
    const float* wg = (const float*)d_in[19], *wga = (const float*)d_in[20], *wgb = (const float*)d_in[21];
    const float* wu = (const float*)d_in[22], *wua = (const float*)d_in[23], *wub = (const float*)d_in[24];
    const float* wd = (const float*)d_in[25], *wda = (const float*)d_in[26], *wdb = (const float*)d_in[27];
    float* out = (float*)d_out;

    char* ws = (char*)d_ws;
    const size_t SZ_DD  = (size_t)Dsz * Dsz * 2;     // 8 MiB bf16 weight
    const size_t SZ_DF  = (size_t)Dsz * FFsz * 2;    // 32 MiB bf16 weight
    const size_t SZ_ACT = (size_t)Msz * Dsz * 2;     // 16 MiB bf16 activations
    const size_t SZ_ACTF= (size_t)Msz * Dsz * 4;     // 32 MiB fp32 activations
    const size_t SZ_FF  = (size_t)Msz * FFsz * 2;    // 64 MiB bf16
    size_t o = 0;
    u16* Wq = (u16*)(ws + o); o += SZ_DD;
    u16* Wk = (u16*)(ws + o); o += SZ_DD;
    u16* Wv = (u16*)(ws + o); o += SZ_DD;
    u16* Wo = (u16*)(ws + o); o += SZ_DD;
    u16* Wg = (u16*)(ws + o); o += SZ_DF;
    u16* Wu = (u16*)(ws + o); o += SZ_DF;
    u16* Wd = (u16*)(ws + o); o += SZ_DF;
    u16* h1 = (u16*)(ws + o); o += SZ_ACT;
    u16* qr = (u16*)(ws + o); o += SZ_ACT;
    u16* kr = (u16*)(ws + o); o += SZ_ACT;
    u16* vb = (u16*)(ws + o); o += SZ_ACT;
    u16* qrot = (u16*)(ws + o); o += SZ_ACT;
    u16* krot = (u16*)(ws + o); o += SZ_ACT;
    u16* attn = (u16*)(ws + o); o += SZ_ACT;
    float* x1 = (float*)(ws + o); o += SZ_ACTF;
    u16* h2 = (u16*)(ws + o); o += SZ_ACT;
    u16* gate = (u16*)(ws + o); o += SZ_FF;
    u16* had  = (u16*)(ws + o); o += SZ_FF;

    // --- 1. fold LoRA into weights, convert to bf16, transpose ---
    prep_weight_kernel<<<dim3(Dsz/32,  Dsz/32), 256, 0, stream>>>(wq, wqa, wqb, Wq, Dsz, Dsz);
    prep_weight_kernel<<<dim3(Dsz/32,  Dsz/32), 256, 0, stream>>>(wk, wka, wkb, Wk, Dsz, Dsz);
    prep_weight_kernel<<<dim3(Dsz/32,  Dsz/32), 256, 0, stream>>>(wv, wva, wvb, Wv, Dsz, Dsz);
    prep_weight_kernel<<<dim3(Dsz/32,  Dsz/32), 256, 0, stream>>>(wo, woa, wob, Wo, Dsz, Dsz);
    prep_weight_kernel<<<dim3(FFsz/32, Dsz/32), 256, 0, stream>>>(wg, wga, wgb, Wg, Dsz, FFsz);
    prep_weight_kernel<<<dim3(FFsz/32, Dsz/32), 256, 0, stream>>>(wu, wua, wub, Wu, Dsz, FFsz);
    prep_weight_kernel<<<dim3(Dsz/32, FFsz/32), 256, 0, stream>>>(wd, wda, wdb, Wd, FFsz, Dsz);

    // --- 2. attention block ---
    rmsnorm_kernel<<<Msz, 256, 0, stream>>>(x, nw1, h1);
    gemm_bf16_kernel<2><<<dim3(Dsz/128, Msz/128), 256, 0, stream>>>(h1, Wq, nullptr, qr, nullptr, nullptr, Msz, Dsz, Dsz);
    gemm_bf16_kernel<2><<<dim3(Dsz/128, Msz/128), 256, 0, stream>>>(h1, Wk, nullptr, kr, nullptr, nullptr, Msz, Dsz, Dsz);
    gemm_bf16_kernel<2><<<dim3(Dsz/128, Msz/128), 256, 0, stream>>>(h1, Wv, nullptr, vb, nullptr, nullptr, Msz, Dsz, Dsz);
    rope_kernel<<<(Msz * Dsz) / 256, 256, 0, stream>>>(qr, qrot, cosT, sinT, pos);
    rope_kernel<<<(Msz * Dsz) / 256, 256, 0, stream>>>(kr, krot, cosT, sinT, pos);
    flash_attn_kernel<<<dim3(Ssz/64, Hsz, Bsz), 128, 0, stream>>>(qrot, krot, vb, attn);
    gemm_bf16_kernel<1><<<dim3(Dsz/128, Msz/128), 256, 0, stream>>>(attn, Wo, x1, nullptr, x, nullptr, Msz, Dsz, Dsz);

    // --- 3. gated MLP block ---
    rmsnorm_kernel<<<Msz, 256, 0, stream>>>(x1, nw2, h2);
    gemm_bf16_kernel<2><<<dim3(FFsz/128, Msz/128), 256, 0, stream>>>(h2, Wg, nullptr, gate, nullptr, nullptr, Msz, FFsz, Dsz);
    gemm_bf16_kernel<3><<<dim3(FFsz/128, Msz/128), 256, 0, stream>>>(h2, Wu, nullptr, had, nullptr, gate, Msz, FFsz, Dsz);
    gemm_bf16_kernel<1><<<dim3(Dsz/128, Msz/128), 256, 0, stream>>>(had, Wd, out, nullptr, x1, nullptr, Msz, Dsz, FFsz);
}
